// MultiHeadAttention_52226802319516
// MI455X (gfx1250) — matrix operand order repaired and checked by reading, not compiled, not run on hardware
//
#include <hip/hip_runtime.h>
#include <hip/hip_bf16.h>

// ---------------------------------------------------------------------------
// MHA forward for MI455X (gfx1250, wave32, WMMA).
// bf16 WMMA everywhere, fp32 accumulation.  CDNA5 data movement:
//  - ctx GEMM: V tiles staged in LDS via global_load_async_to_lds_b128
//    (double buffered, s_wait_asynccnt).
//  - scores GEMM: Q tile staged per-block by the Tensor Data Mover
//    (tensor_load_to_lds + s_wait_tensorcnt).
// ---------------------------------------------------------------------------

typedef __attribute__((ext_vector_type(16))) __bf16 bf16x16;
typedef __attribute__((ext_vector_type(8)))  float  f32x8;
typedef int v4i __attribute__((vector_size(16)));   // matches builtin's V4i

#define BS     4
#define SEQ    2048
#define DM     512
#define NH     8
#define DK     64
#define SCALE  0.125f          // 1/sqrt(64)
#define NEGBIG (-1.0e10f)

#if __has_builtin(__builtin_amdgcn_global_load_async_to_lds_b128) && \
    __has_builtin(__builtin_amdgcn_s_wait_asynccnt)
#define HAVE_ASYNC_LDS 1
#else
#define HAVE_ASYNC_LDS 0
#endif

#if __has_builtin(__builtin_amdgcn_tensor_load_to_lds) && \
    __has_builtin(__builtin_amdgcn_s_wait_tensorcnt)
#define HAVE_TDM 1
#else
#define HAVE_TDM 0
#endif

#define AS1 __attribute__((address_space(1)))
#define AS3 __attribute__((address_space(3)))

// ---- fragment loaders (ISA 7.12.2 layouts, wave32) -------------------------
// A (16x32, 16-bit): lanes 0-15 -> M=0..15 (K 0-7 in V0..3, 16-23 in V4..7),
//                    lanes 16-31 -> M=0..15 (K 8-15 in V0..3, 24-31 in V4..7).
template <typename T>
__device__ __forceinline__ bf16x16
load_a_frag(const T* __restrict__ A, int lda, int m0, int k0, int lane) {
  bf16x16 a;
  const int m = m0 + (lane & 15);
  const int g = lane >> 4;
  const T* row = A + (size_t)m * lda;
#pragma unroll
  for (int v = 0; v < 8; ++v) {
    const int k = k0 + g * 8 + (v & 3) * 2 + (v >> 2) * 16;
    a[2 * v]     = (__bf16)row[k];
    a[2 * v + 1] = (__bf16)row[k + 1];
  }
  return a;
}

// B (32x16, 16-bit), where B[k][n] = W[n][k]  (multiplying by W^T):
// lanes 0-15 -> K 0-7,16-23; lanes 16-31 -> K 8-15,24-31; 2 consecutive K per VGPR.
template <typename T>
__device__ __forceinline__ bf16x16
load_bT_frag(const T* __restrict__ W, int ldw, int n0, int k0, int lane) {
  bf16x16 b;
  const int n = n0 + (lane & 15);
  const int g = lane >> 4;
  const T* row = W + (size_t)n * ldw;
#pragma unroll
  for (int v = 0; v < 8; ++v) {
    const int k = k0 + g * 8 + (v & 3) * 2 + (v >> 2) * 16;
    b[2 * v]     = (__bf16)row[k];
    b[2 * v + 1] = (__bf16)row[k + 1];
  }
  return b;
}

// B (32x16, 16-bit) from an LDS tile laid out [32][ldb] row-major (bf16).
__device__ __forceinline__ bf16x16
load_b_lds(const __bf16* tile, int ldb, int n0, int lane) {
  bf16x16 b;
  const int n = n0 + (lane & 15);
  const int g = lane >> 4;
#pragma unroll
  for (int v = 0; v < 8; ++v) {
    const int k = g * 8 + (v & 3) * 2 + (v >> 2) * 16;
    b[2 * v]     = tile[k * ldb + n];
    b[2 * v + 1] = tile[(k + 1) * ldb + n];
  }
  return b;
}

// A (16x32) from an LDS tile [16][lda] row-major (bf16).
__device__ __forceinline__ bf16x16
load_a_lds(const __bf16* tile, int lda, int k0, int lane) {
  bf16x16 a;
  const int m = lane & 15;
  const int g = lane >> 4;
  const __bf16* row = tile + m * lda;
#pragma unroll
  for (int v = 0; v < 8; ++v) {
    const int k = k0 + g * 8 + (v & 3) * 2 + (v >> 2) * 16;
    a[2 * v]     = row[k];
    a[2 * v + 1] = row[k + 1];
  }
  return a;
}

__device__ __forceinline__ f32x8 wmma_bf16(bf16x16 a, bf16x16 b, f32x8 c) {
  return __builtin_amdgcn_wmma_f32_16x16x32_bf16(
      /*neg_a=*/false, a, /*neg_b=*/false, b,
      /*c_mod=*/(short)0, c, /*reuse_a=*/false, /*reuse_b=*/false);
}

#if HAVE_TDM
typedef __attribute__((ext_vector_type(4))) unsigned int u32x4;
typedef __attribute__((ext_vector_type(8))) int          i32x8;
typedef __attribute__((ext_vector_type(4))) int          i32x4;

// 2D bf16 tile load via Tensor Data Mover.  D# per ISA 8.3/8.4:
// group0: count=1, lds_addr, 57b global_addr, type=2.
// group1: data_size=2B, tensor_dim0/1, tile_dim0/1, tensor_dim0_stride.
__device__ __forceinline__ void
tdm_load_tile_2d(unsigned lds_off, const void* gaddr,
                 unsigned tensor_d0, unsigned tensor_d1,
                 unsigned tile_d0, unsigned tile_d1, unsigned stride0) {
  const unsigned long long ga = (unsigned long long)(uintptr_t)gaddr;
  u32x4 g0;
  g0[0] = 1u;                                   // count=1 (valid), user mode
  g0[1] = lds_off;                              // LDS byte address
  g0[2] = (unsigned)ga;                         // global_addr[31:0]
  g0[3] = (unsigned)(ga >> 32) | (2u << 30);    // global_addr[56:32] | type=2
  i32x8 g1;
  g1[0] = (int)(1u << 16);                      // data_size=1 (2 bytes)
  g1[1] = (int)((tensor_d0 & 0xFFFFu) << 16);   // tensor_dim0[15:0]
  g1[2] = (int)((tensor_d0 >> 16) | ((tensor_d1 & 0xFFFFu) << 16));
  g1[3] = (int)((tensor_d1 >> 16) | (tile_d0 << 16));
  g1[4] = (int)(tile_d1 & 0xFFFFu);             // tile_dim1 | tile_dim2=0
  g1[5] = (int)stride0;                         // tensor_dim0_stride[31:0]
  g1[6] = 0;
  g1[7] = 0;
  i32x4 z4 = {0, 0, 0, 0};
#if defined(__clang_major__) && __clang_major__ >= 23
  i32x8 z8 = {0, 0, 0, 0, 0, 0, 0, 0};
  __builtin_amdgcn_tensor_load_to_lds(g0, g1, z4, z4, z8, 0);
#else
  __builtin_amdgcn_tensor_load_to_lds(g0, g1, z4, z4, 0);
#endif
}
#endif  // HAVE_TDM

// ---------------------------------------------------------------------------
// Kernel 1: projection  outH[b,h,s,dk] = (X @ W^T + bias)  (bf16 head layout)
// grid(4,512) block(256): 8 waves, each wave one 16x16 tile.
// ---------------------------------------------------------------------------
__global__ void __launch_bounds__(256)
proj_kernel(const float* __restrict__ X, const float* __restrict__ W,
            const float* __restrict__ bias, __bf16* __restrict__ outH) {
  const int lane = threadIdx.x & 31;
  const int wave = threadIdx.x >> 5;
  const int n0 = (blockIdx.x * 8 + wave) * 16;
  const int m0 = blockIdx.y * 16;

  f32x8 c = {};
#pragma unroll 4
  for (int k0 = 0; k0 < DM; k0 += 32) {
    bf16x16 a = load_a_frag(X, DM, m0, k0, lane);
    bf16x16 b = load_bT_frag(W, DM, n0, k0, lane);
    c = wmma_bf16(a, b, c);
  }

  const int n  = n0 + (lane & 15);
  const int g  = lane >> 4;
  const int h  = n >> 6;
  const int dk = n & 63;
  const float bb = bias[n];
#pragma unroll
  for (int r = 0; r < 8; ++r) {
    const int m  = m0 + r + g * 8;
    const int bi = m >> 11;
    const int s  = m & 2047;
    outH[(((size_t)(bi * NH + h)) * SEQ + s) * DK + dk] = (__bf16)(c[r] + bb);
  }
}

// ---------------------------------------------------------------------------
// Kernel 2: masked scaled scores.  attn[bh,q,k] = mask ? 0.125*Q.K : -1e10
// grid(16,128,32) block(256).  Q tile (16x64 bf16, shared by all 8 waves)
// staged in LDS by the Tensor Data Mover.
// ---------------------------------------------------------------------------
__global__ void __launch_bounds__(256)
scores_kernel(const __bf16* __restrict__ Qh, const __bf16* __restrict__ Kh,
              const int* __restrict__ mask, float* __restrict__ attn) {
  __shared__ __bf16 qtile[16 * DK];          // 2 KB
  const int lane = threadIdx.x & 31;
  const int wave = threadIdx.x >> 5;
  const int tid  = threadIdx.x;
  const int bh = blockIdx.z;
  const int b  = bh >> 3;
  const int q0 = blockIdx.y * 16;
  const int n0 = (blockIdx.x * 8 + wave) * 16;

  const __bf16* Q = Qh + (size_t)bh * SEQ * DK;
  const __bf16* K = Kh + (size_t)bh * SEQ * DK;

#if HAVE_TDM
  if (wave == 0) {
    const unsigned lds_off = (unsigned)(size_t)(AS3 __bf16*)(__bf16*)qtile;
    tdm_load_tile_2d(lds_off, Q + (size_t)q0 * DK,
                     /*tensor_d0=*/DK, /*tensor_d1=*/SEQ,
                     /*tile_d0=*/DK, /*tile_d1=*/16, /*stride0=*/DK);
    __builtin_amdgcn_s_wait_tensorcnt(0);
  }
#else
  for (int i = tid; i < 16 * DK; i += 256) qtile[i] = Q[(size_t)q0 * DK + i];
#endif
  __syncthreads();

  f32x8 c = {};
#pragma unroll
  for (int kk = 0; kk < DK; kk += 32) {
    bf16x16 a   = load_a_lds(qtile, DK, kk, lane);
    bf16x16 bfr = load_bT_frag(K, DK, n0, kk, lane);  // B[k][n] = Kh[n][k]
    c = wmma_bf16(a, bfr, c);
  }

  const int n = n0 + (lane & 15);
  const int g = lane >> 4;
  float* out = attn + (size_t)bh * SEQ * SEQ;
#pragma unroll
  for (int r = 0; r < 8; ++r) {
    const int q = q0 + r + g * 8;
    const int mv = mask[((size_t)b * SEQ + q) * SEQ + n];
    out[(size_t)q * SEQ + n] = (mv == 0) ? NEGBIG : c[r] * SCALE;
  }
}

// ---------------------------------------------------------------------------
// Kernel 3: in-place row softmax over sk=2048.  One block per row.
// ---------------------------------------------------------------------------
__global__ void __launch_bounds__(256) softmax_kernel(float* __restrict__ attn) {
  __shared__ float red[256];
  float* row = attn + (size_t)blockIdx.x * SEQ;
  const int tid = threadIdx.x;

  float mx = -3.4e38f;
  for (int i = tid; i < SEQ; i += 256) mx = fmaxf(mx, row[i]);
  red[tid] = mx;
  __syncthreads();
  for (int s = 128; s > 0; s >>= 1) {
    if (tid < s) red[tid] = fmaxf(red[tid], red[tid + s]);
    __syncthreads();
  }
  mx = red[0];
  __syncthreads();

  float sum = 0.0f;
  for (int i = tid; i < SEQ; i += 256) {
    const float e = __expf(row[i] - mx);
    row[i] = e;
    sum += e;
  }
  red[tid] = sum;
  __syncthreads();
  for (int s = 128; s > 0; s >>= 1) {
    if (tid < s) red[tid] += red[tid + s];
    __syncthreads();
  }
  const float inv = 1.0f / red[0];
  for (int i = tid; i < SEQ; i += 256) row[i] *= inv;
}

// ---------------------------------------------------------------------------
// Kernel 4: ctx = attn @ Vh -> concat layout [b, s, h*64+d] (bf16).
// Per (b,h): M=2048, N=64, K=2048.  grid(16,32) block(256).
// Block tile: 128(M) x 64(N); wave w owns rows m0+16w, 4 accumulators.
// V k-slab (32x64 bf16 = 4 KB) double-buffered in LDS via async copies.
// ---------------------------------------------------------------------------
__global__ void __launch_bounds__(256)
ctx_kernel(const float* __restrict__ attn, const __bf16* __restrict__ Vh,
           __bf16* __restrict__ concat) {
  __shared__ __bf16 vtile[2 * 32 * DK];      // 2 x 4 KB
  const int lane = threadIdx.x & 31;
  const int wave = threadIdx.x >> 5;
  const int tid  = threadIdx.x;
  const int bh = blockIdx.y;
  const int b  = bh >> 3;
  const int h  = bh & 7;
  const int m0 = blockIdx.x * 128 + wave * 16;

  const float*  A = attn + (size_t)bh * SEQ * SEQ;
  const __bf16* V = Vh + (size_t)bh * SEQ * DK;

  // each thread moves one 16B chunk of the 32x64 bf16 slab
  const int vrow = tid >> 3;            // 0..31
  const int vcol = (tid & 7) * 8;       // bf16 element column

  auto fill = [&](int buf, int kk) {
    const __bf16* src = V + (size_t)(kk + vrow) * DK + vcol;
    __bf16* dst = &vtile[buf * (32 * DK) + vrow * DK + vcol];
#if HAVE_ASYNC_LDS
    __builtin_amdgcn_global_load_async_to_lds_b128(
        (AS1 v4i*)(uintptr_t)src, (AS3 v4i*)(AS3 void*)dst, 0, 0);
#else
#pragma unroll
    for (int j = 0; j < 8; ++j) dst[j] = src[j];
#endif
  };

  f32x8 acc[4] = {{}, {}, {}, {}};
  fill(0, 0);

  for (int i = 0; i < SEQ / 32; ++i) {
    const int kk  = i * 32;
    const int cur = i & 1;
    if (kk + 32 < SEQ) {
      fill(1 - cur, kk + 32);
#if HAVE_ASYNC_LDS
      __builtin_amdgcn_s_wait_asynccnt(1);   // cur slab done (in-order)
#endif
    } else {
#if HAVE_ASYNC_LDS
      __builtin_amdgcn_s_wait_asynccnt(0);
#endif
    }
    __syncthreads();

    // prefetch upcoming attn rows (gfx1250 global_prefetch)
    if (kk + 64 < SEQ)
      __builtin_prefetch(&A[(size_t)(m0 + (lane & 15)) * SEQ + kk + 64], 0, 3);

    bf16x16 a = load_a_frag(A, SEQ, m0, kk, lane);
    const __bf16* slab = &vtile[cur * (32 * DK)];
#pragma unroll
    for (int nt = 0; nt < 4; ++nt)
      acc[nt] = wmma_bf16(a, load_b_lds(slab, DK, nt * 16, lane), acc[nt]);

    __syncthreads();
  }

  const int g = lane >> 4;
#pragma unroll
  for (int nt = 0; nt < 4; ++nt) {
    const int n = nt * 16 + (lane & 15);
#pragma unroll
    for (int r = 0; r < 8; ++r) {
      const int s = m0 + r + g * 8;
      concat[((size_t)(b * SEQ + s)) * DM + h * DK + n] = (__bf16)acc[nt][r];
    }
  }
}

// ---------------------------------------------------------------------------
// Kernel 5: output projection  out = concat @ Wo^T + bo  (fp32 out).
// ---------------------------------------------------------------------------
__global__ void __launch_bounds__(256)
outproj_kernel(const __bf16* __restrict__ concat, const float* __restrict__ Wo,
               const float* __restrict__ bo, float* __restrict__ out) {
  const int lane = threadIdx.x & 31;
  const int wave = threadIdx.x >> 5;
  const int n0 = (blockIdx.x * 8 + wave) * 16;
  const int m0 = blockIdx.y * 16;

  f32x8 c = {};
#pragma unroll 4
  for (int k0 = 0; k0 < DM; k0 += 32) {
    bf16x16 a = load_a_frag(concat, DM, m0, k0, lane);
    bf16x16 b = load_bT_frag(Wo, DM, n0, k0, lane);
    c = wmma_bf16(a, b, c);
  }

  const int n = n0 + (lane & 15);
  const int g = lane >> 4;
  const float bb = bo[n];
#pragma unroll
  for (int r = 0; r < 8; ++r) {
    const int m = m0 + r + g * 8;
    out[(size_t)m * DM + n] = c[r] + bb;
  }
}

// ---------------------------------------------------------------------------
extern "C" void kernel_launch(void* const* d_in, const int* in_sizes, int n_in,
                              void* d_out, int out_size, void* d_ws, size_t ws_size,
                              hipStream_t stream) {
  const float* q    = (const float*)d_in[0];
  const float* k    = (const float*)d_in[1];
  const float* v    = (const float*)d_in[2];
  const int*   mask = (const int*)d_in[3];
  const float* Wq   = (const float*)d_in[4];
  const float* bq   = (const float*)d_in[5];
  const float* Wk   = (const float*)d_in[6];
  const float* bk   = (const float*)d_in[7];
  const float* Wv   = (const float*)d_in[8];
  const float* bv   = (const float*)d_in[9];
  const float* Wo   = (const float*)d_in[10];
  const float* bo   = (const float*)d_in[11];

  float* out  = (float*)d_out;                      // [4,2048,512]
  float* attn = out + (size_t)BS * SEQ * DM;        // [4,8,2048,2048]

  const size_t headElems = (size_t)BS * NH * SEQ * DK;
  __bf16* Qh     = (__bf16*)d_ws;
  __bf16* Kh     = Qh + headElems;
  __bf16* Vh     = Kh + headElems;
  __bf16* concat = Vh + headElems;                  // [8192, 512] bf16

  const dim3 blk(256);

  proj_kernel<<<dim3(4, 512), blk, 0, stream>>>(q, Wq, bq, Qh);
  proj_kernel<<<dim3(4, 512), blk, 0, stream>>>(k, Wk, bk, Kh);
  proj_kernel<<<dim3(4, 512), blk, 0, stream>>>(v, Wv, bv, Vh);

  scores_kernel<<<dim3(16, 128, 32), blk, 0, stream>>>(Qh, Kh, mask, attn);

  softmax_kernel<<<dim3(BS * NH * SEQ), blk, 0, stream>>>(attn);

  ctx_kernel<<<dim3(16, 32), blk, 0, stream>>>(attn, Vh, concat);

  outproj_kernel<<<dim3(4, 512), blk, 0, stream>>>(concat, Wo, bo, out);
}
